// mLSTMCell_27169963115051
// MI455X (gfx1250) — matrix operand order repaired and checked by reading, not compiled, not run on hardware
//
#include <hip/hip_runtime.h>
#include <hip/hip_bf16.h>

// ---------------------------------------------------------------------------
// mLSTM (xLSTM) for MI455X / gfx1250.
// Phase 1: time-parallel projections as one bf16 WMMA GEMM (split-bf16 for
//          ~fp32 accuracy, K tripled to 3072), LDS staged via CDNA5
//          GLOBAL_LOAD_ASYNC_TO_LDS (ASYNCcnt) when available.
// Phase 2: sequential per-head scan, c-state in VGPRs, software-pipelined
//          projection loads + global_prefetch for t+2.
// ---------------------------------------------------------------------------

#define T_STEPS 8192
#define D_IN    1024
#define H_HEADS 16
#define HEAD    64
#define RPH     264                    // rows per head (o:64,k:64,q:64,v:64,i,f,pad6)
#define ROWSP   (H_HEADS * RPH)        // 4224, multiple of 128
#define DK      (3 * D_IN)             // 3072: [hi|hi|lo] x [hi|lo|hi] split-bf16
#define BM      128
#define BN      128
#define BK      32

typedef __bf16 v16bf __attribute__((ext_vector_type(16)));
typedef float  v8f   __attribute__((ext_vector_type(8)));
typedef int    v4i   __attribute__((ext_vector_type(4)));

#if defined(__has_builtin)
#if __has_builtin(__builtin_amdgcn_global_load_async_to_lds_b128)
#define USE_ASYNC_LDS 1
#endif
#endif
#ifndef USE_ASYNC_LDS
#define USE_ASYNC_LDS 0
#endif

__device__ __forceinline__ unsigned short f2bf(float f) {
    union { float f; unsigned u; } v; v.f = f;
    unsigned u = v.u;
    return (unsigned short)((u + 0x7FFFu + ((u >> 16) & 1u)) >> 16); // RNE
}
__device__ __forceinline__ float bf2f(unsigned short b) {
    union { unsigned u; float f; } v; v.u = ((unsigned)b) << 16;
    return v.f;
}

#if USE_ASYNC_LDS
// prototype (from hipcc diagnostic): (v4i AS1*, v4i AS3*, imm i32 offset, imm i32 cpol)
typedef __attribute__((address_space(1))) v4i* as1_v4i_ptr;
typedef __attribute__((address_space(3))) v4i* as3_v4i_ptr;

__device__ __forceinline__ void async_cp16(const unsigned short* g, unsigned short* l) {
    __builtin_amdgcn_global_load_async_to_lds_b128(
        (as1_v4i_ptr)(g), (as3_v4i_ptr)(l), 0, 0);
}
__device__ __forceinline__ void wait_async0() {
#if __has_builtin(__builtin_amdgcn_s_wait_asynccnt)
    __builtin_amdgcn_s_wait_asynccnt(0);
#else
    asm volatile("s_wait_asynccnt 0x0" ::: "memory");
#endif
}
#endif

// ------------------------- pack weights (fp32 -> split bf16) ----------------
__global__ void pack_weights(const float* __restrict__ ifw,   // (H,2,D)
                             const float* __restrict__ ow,    // (H,64,D)
                             const float* __restrict__ kqw,   // (H,128,D)
                             const float* __restrict__ vw,    // (H,64,D)
                             unsigned short* __restrict__ Wp) // (ROWSP, DK)
{
    long idx = (long)blockIdx.x * blockDim.x + threadIdx.x;
    if (idx >= (long)ROWSP * D_IN) return;
    int r = (int)(idx / D_IN);
    int d = (int)(idx - (long)r * D_IN);
    int h  = r / RPH;
    int lr = r - h * RPH;

    float w = 0.0f;
    if      (lr < 64)   w = ow [((long)(h *  64 +  lr        )) * D_IN + d];
    else if (lr < 128)  w = kqw[((long)(h * 128 + (lr -  64) )) * D_IN + d]; // k rows
    else if (lr < 192)  w = kqw[((long)(h * 128 + 64 + (lr - 128))) * D_IN + d]; // q rows
    else if (lr < 256)  w = vw [((long)(h *  64 + (lr - 192) )) * D_IN + d];
    else if (lr == 256) w = ifw[((long)(h * 2 + 0)) * D_IN + d];             // i
    else if (lr == 257) w = ifw[((long)(h * 2 + 1)) * D_IN + d];             // f
    // lr in [258,264): zero padding

    unsigned short hi = f2bf(w);
    unsigned short lo = f2bf(w - bf2f(hi));
    unsigned short* row = Wp + (size_t)r * DK;
    row[d]              = hi;   // term Whi*xhi
    row[D_IN + d]       = hi;   // term Whi*xlo
    row[2 * D_IN + d]   = lo;   // term Wlo*xhi
}

// ------------------------- pack xs (fp32 -> split bf16) ---------------------
__global__ void pack_x(const float* __restrict__ xs,           // (T, D)
                       unsigned short* __restrict__ Xb)        // (T, DK)
{
    long idx = (long)blockIdx.x * blockDim.x + threadIdx.x;
    if (idx >= (long)T_STEPS * D_IN) return;
    int t = (int)(idx / D_IN);
    int d = (int)(idx - (long)t * D_IN);
    float x = xs[idx];
    unsigned short hi = f2bf(x);
    unsigned short lo = f2bf(x - bf2f(hi));
    unsigned short* row = Xb + (size_t)t * DK;
    row[d]            = hi;
    row[D_IN + d]     = lo;
    row[2 * D_IN + d] = hi;
}

// ------------------------- WMMA projection GEMM -----------------------------
// P[t, r] = sum_k Wp[r, k] * Xb[t, k]   (f32 accumulate)
__launch_bounds__(256)
__global__ void proj_gemm(const unsigned short* __restrict__ Wp,
                          const unsigned short* __restrict__ Xb,
                          float* __restrict__ P)
{
    __shared__ unsigned short As[2][BM * BK];   // 8 KB each
    __shared__ unsigned short Bs[2][BN * BK];   // 8 KB each

    const int tid  = threadIdx.x;
    const int wave = tid >> 5;        // 0..7
    const int lane = tid & 31;
    const int wm   = wave >> 1;       // 0..3 -> M offset wm*32
    const int wn   = wave & 1;        // 0..1 -> N offset wn*64
    const int mBase = blockIdx.y * BM;
    const int nBase = blockIdx.x * BN;

    // cooperative global->LDS staging indices: 2 threads per row, 16 elems each
    const int lr   = tid >> 1;        // 0..127
    const int lseg = (tid & 1) << 4;  // 0 or 16

    const unsigned short* gA = Wp + (size_t)(mBase + lr) * DK + lseg;
    const unsigned short* gB = Xb + (size_t)(nBase + lr) * DK + lseg;

    v8f acc[2][4];
    #pragma unroll
    for (int mt = 0; mt < 2; ++mt)
        #pragma unroll
        for (int nt = 0; nt < 4; ++nt)
            acc[mt][nt] = {};

    const int l15 = lane & 15;
    const int kbA = (lane >> 4) * 16;   // A: contiguous K-16 per lane-half
    const int kbB = (lane >> 4) * 16;   // B: contiguous K-16 per lane-half
    const int nIter = DK / BK;          // 96

#if USE_ASYNC_LDS
    // ---- CDNA5 async global->LDS staging, ASYNCcnt-tracked -----------------
    {   // prologue: slice 0 -> buffer 0
        async_cp16(gA,     &As[0][lr * BK + lseg]);
        async_cp16(gA + 8, &As[0][lr * BK + lseg + 8]);
        async_cp16(gB,     &Bs[0][lr * BK + lseg]);
        async_cp16(gB + 8, &Bs[0][lr * BK + lseg + 8]);
    }
    wait_async0();
    __syncthreads();

    for (int it = 0; it < nIter; ++it) {
        const int cur = it & 1;
        const bool hasNext = (it + 1) < nIter;
        if (hasNext) {
            // buffer cur^1: all reads of it finished before the previous barrier
            const int nxt = cur ^ 1;
            const unsigned short* pa = gA + (size_t)(it + 1) * BK;
            const unsigned short* pb = gB + (size_t)(it + 1) * BK;
            async_cp16(pa,     &As[nxt][lr * BK + lseg]);
            async_cp16(pa + 8, &As[nxt][lr * BK + lseg + 8]);
            async_cp16(pb,     &Bs[nxt][lr * BK + lseg]);
            async_cp16(pb + 8, &Bs[nxt][lr * BK + lseg + 8]);
        }

        union AF { v16bf v; uint4 q[2]; } af[2];
        v16bf bfr[4];
        #pragma unroll
        for (int mt = 0; mt < 2; ++mt) {
            const int row = wm * 32 + mt * 16 + l15;
            af[mt].q[0] = *(const uint4*)&As[cur][row * BK + kbA];
            af[mt].q[1] = *(const uint4*)&As[cur][row * BK + kbA + 8];
        }
        #pragma unroll
        for (int nt = 0; nt < 4; ++nt) {
            const int col = wn * 64 + nt * 16 + l15;
            bfr[nt] = *(const v16bf*)&Bs[cur][col * BK + kbB];
        }
        #pragma unroll
        for (int mt = 0; mt < 2; ++mt)
            #pragma unroll
            for (int nt = 0; nt < 4; ++nt)
                acc[mt][nt] = __builtin_amdgcn_wmma_f32_16x16x32_bf16(
                    false, af[mt].v, false, bfr[nt],
                    (short)0, acc[mt][nt], false, false);

        if (hasNext) wait_async0();   // next slice landed in LDS (this wave)
        __syncthreads();              // all waves: reads of cur done + data visible
    }
#else
    // ---- fallback: register-staged global->LDS -----------------------------
    {
        uint4 a0 = *(const uint4*)(gA);
        uint4 a1 = *(const uint4*)(gA + 8);
        uint4 b0 = *(const uint4*)(gB);
        uint4 b1 = *(const uint4*)(gB + 8);
        *(uint4*)&As[0][lr * BK + lseg]     = a0;
        *(uint4*)&As[0][lr * BK + lseg + 8] = a1;
        *(uint4*)&Bs[0][lr * BK + lseg]     = b0;
        *(uint4*)&Bs[0][lr * BK + lseg + 8] = b1;
    }
    __syncthreads();

    for (int it = 0; it < nIter; ++it) {
        const int cur = it & 1;
        const bool hasNext = (it + 1) < nIter;

        uint4 a0, a1, b0, b1;
        if (hasNext) {
            const unsigned short* pa = gA + (size_t)(it + 1) * BK;
            const unsigned short* pb = gB + (size_t)(it + 1) * BK;
            a0 = *(const uint4*)(pa);
            a1 = *(const uint4*)(pa + 8);
            b0 = *(const uint4*)(pb);
            b1 = *(const uint4*)(pb + 8);
        }

        union AF { v16bf v; uint4 q[2]; } af[2];
        v16bf bfr[4];
        #pragma unroll
        for (int mt = 0; mt < 2; ++mt) {
            const int row = wm * 32 + mt * 16 + l15;
            af[mt].q[0] = *(const uint4*)&As[cur][row * BK + kbA];
            af[mt].q[1] = *(const uint4*)&As[cur][row * BK + kbA + 8];
        }
        #pragma unroll
        for (int nt = 0; nt < 4; ++nt) {
            const int col = wn * 64 + nt * 16 + l15;
            bfr[nt] = *(const v16bf*)&Bs[cur][col * BK + kbB];
        }
        #pragma unroll
        for (int mt = 0; mt < 2; ++mt)
            #pragma unroll
            for (int nt = 0; nt < 4; ++nt)
                acc[mt][nt] = __builtin_amdgcn_wmma_f32_16x16x32_bf16(
                    false, af[mt].v, false, bfr[nt],
                    (short)0, acc[mt][nt], false, false);

        if (hasNext) {
            const int nxt = cur ^ 1;
            *(uint4*)&As[nxt][lr * BK + lseg]     = a0;
            *(uint4*)&As[nxt][lr * BK + lseg + 8] = a1;
            *(uint4*)&Bs[nxt][lr * BK + lseg]     = b0;
            *(uint4*)&Bs[nxt][lr * BK + lseg + 8] = b1;
        }
        __syncthreads();
    }
#endif

    // epilogue: C/D layout -> P[t, r] (P stored (T, ROWSP))
    const int rhi = lane >> 4;
    #pragma unroll
    for (int mt = 0; mt < 2; ++mt) {
        #pragma unroll
        for (int nt = 0; nt < 4; ++nt) {
            const int n = nBase + wn * 64 + nt * 16 + l15;
            #pragma unroll
            for (int r = 0; r < 8; ++r) {
                const int m = mBase + wm * 32 + mt * 16 + 8 * rhi + r;
                P[(size_t)n * ROWSP + m] = acc[mt][nt][r];
            }
        }
    }
}

// ------------------------- sequential mLSTM scan ----------------------------
// one block per head, 64 threads; thread x owns row x of c (64 VGPRs);
// t+1 projections register-double-buffered, t+2 prefetched.
__launch_bounds__(64)
__global__ void mlstm_scan(const float* __restrict__ P,        // (T, ROWSP)
                           const float* __restrict__ if_bias,  // (H,2)
                           const float* __restrict__ o_bias,   // (H,64)
                           const float* __restrict__ kvq_bias, // (H,3,64)
                           float* __restrict__ out)            // (T, H*64)
{
    const int h = blockIdx.x;
    const int x = threadIdx.x;

    __shared__ float sk[HEAD], sq[HEAD], snq[HEAD];

    float c[HEAD];
    #pragma unroll
    for (int j = 0; j < HEAD; ++j) c[j] = 0.0f;
    float nx = 0.0f, mprev = 0.0f;

    const float ob = o_bias[h * HEAD + x];
    const float kb = kvq_bias[(h * 3 + 0) * HEAD + x];
    const float vb = kvq_bias[(h * 3 + 1) * HEAD + x];
    const float qb = kvq_bias[(h * 3 + 2) * HEAD + x];
    const float ib = if_bias[h * 2 + 0];
    const float fb = if_bias[h * 2 + 1];
    const float inv_sqrt_head = 0.125f;   // 1/sqrt(64)

    const float* base = P + (size_t)h * RPH;

    // software pipeline: L_* holds step-t row, N_* prefetches step-(t+1)
    float L_o = base[x];
    float L_k = base[64 + x];
    float L_q = base[128 + x];
    float L_v = base[192 + x];
    float L_i = base[256];
    float L_f = base[257];

    for (int t = 0; t < T_STEPS; ++t) {
        const int t1 = (t + 1 < T_STEPS) ? (t + 1) : t;
        const int t2 = (t + 2 < T_STEPS) ? (t + 2) : t;
        const float* pn = base + (size_t)t1 * ROWSP;
        // issue next-step loads early: they overlap the step-t arithmetic
        const float N_o = pn[x];
        const float N_k = pn[64 + x];
        const float N_q = pn[128 + x];
        const float N_v = pn[192 + x];
        const float N_i = pn[256];
        const float N_f = pn[257];
        // hint t+2 row into cache (global_prefetch_b8); lanes stride 4 floats
        __builtin_prefetch(base + (size_t)t2 * ROWSP + 4 * x, 0, 1);

        const float opre = L_o;
        const float kx   = L_k * inv_sqrt_head + kb;
        const float qx   = L_q + qb;
        const float vx   = L_v + vb;
        const float ipre = L_i + ib;
        const float fpre = L_f + fb;

        const float m  = fmaxf(fpre + mprev, ipre);
        const float ig = __expf(ipre - m);
        const float fg = __expf(fpre + mprev - m);
        mprev = m;

        sk[x] = kx; sq[x] = qx;
        __syncthreads();

        const float iv = ig * vx;
        float hsum = 0.0f;
        #pragma unroll
        for (int j = 0; j < HEAD; ++j) {
            c[j] = fg * c[j] + iv * sk[j];
            hsum += c[j] * sq[j];
        }

        nx = fg * nx + ig * kx;
        snq[x] = nx * qx;
        __syncthreads();

        float dot = 0.0f;
        #pragma unroll
        for (int j = 0; j < HEAD; ++j) dot += snq[j];
        const float denom = fmaxf(fabsf(dot), 1.0f);
        const float og = 1.0f / (1.0f + __expf(-(opre + ob)));

        out[(size_t)t * (H_HEADS * HEAD) + h * HEAD + x] = og * hsum / denom;
        __syncthreads();   // protect sk/sq/snq for next step

        L_o = N_o; L_k = N_k; L_q = N_q; L_v = N_v; L_i = N_i; L_f = N_f;
    }
}

// ------------------------- launch -------------------------------------------
extern "C" void kernel_launch(void* const* d_in, const int* in_sizes, int n_in,
                              void* d_out, int out_size, void* d_ws, size_t ws_size,
                              hipStream_t stream) {
    (void)in_sizes; (void)n_in; (void)out_size; (void)ws_size;

    const float* xs   = (const float*)d_in[0];
    const float* ifw  = (const float*)d_in[1];
    const float* ow   = (const float*)d_in[2];
    const float* kqw  = (const float*)d_in[3];
    const float* vw   = (const float*)d_in[4];
    const float* ifb  = (const float*)d_in[5];
    const float* obi  = (const float*)d_in[6];
    const float* kvqb = (const float*)d_in[7];
    float* out = (float*)d_out;

    // workspace layout
    const size_t WP_BYTES = (size_t)ROWSP * DK * sizeof(unsigned short);   // ~26 MB
    const size_t XB_BYTES = (size_t)T_STEPS * DK * sizeof(unsigned short); // ~50 MB
    unsigned short* Wp = (unsigned short*)d_ws;
    unsigned short* Xb = (unsigned short*)((char*)d_ws + WP_BYTES);
    float*          P  = (float*)((char*)d_ws + WP_BYTES + XB_BYTES);      // ~138 MB

    {
        long n = (long)ROWSP * D_IN;
        pack_weights<<<dim3((unsigned)((n + 255) / 256)), dim3(256), 0, stream>>>(
            ifw, ow, kqw, vw, Wp);
    }
    {
        long n = (long)T_STEPS * D_IN;
        pack_x<<<dim3((unsigned)((n + 255) / 256)), dim3(256), 0, stream>>>(xs, Xb);
    }
    {
        dim3 grid(T_STEPS / BN, ROWSP / BM);   // (64, 33)
        proj_gemm<<<grid, dim3(256), 0, stream>>>(Wp, Xb, P);
    }
    {
        mlstm_scan<<<dim3(H_HEADS), dim3(HEAD), 0, stream>>>(P, ifb, obi, kvqb, out);
    }
}
